// MultiQueryAttention_996432413530
// MI455X (gfx1250) — compile-verified, matrix operand order checked by reading, not run on hardware
//
#include <hip/hip_runtime.h>
#include <hip/hip_bf16.h>

typedef __bf16 bf16_t;
typedef __attribute__((ext_vector_type(16))) __bf16 v16bf;
typedef __attribute__((ext_vector_type(8)))  float  v8f;
typedef __attribute__((ext_vector_type(4)))  unsigned int v4u;
typedef __attribute__((ext_vector_type(8)))  int v8i_t;
typedef __attribute__((ext_vector_type(4)))  int v4i_t;

#define HIDDEN   2048
#define NHEADS   16
#define NKVHEADS 4
#define HEADDIM  128
#define BATCH    2
#define SEQ      2048
#define MROWS    (BATCH * SEQ)   // 4096

union Frag16 { v16bf v; uint4 q[2]; bf16_t h[16]; };

// ---------------------------------------------------------------------------
// TDM: async 2D bf16 tile load Global -> LDS (row-major, no transform).
// D# layout per CDNA5 ISA ch.8: group0 = {flags, lds_addr, global_addr, type},
// group1 = {mask/data_size, tensor_dim0/1, tile_dim0/1/2, strides}.
// Issued by one wave; completion via s_wait_tensorcnt.
// ---------------------------------------------------------------------------
#if __has_builtin(__builtin_amdgcn_tensor_load_to_lds)
#define HAVE_TDM 1
__device__ __forceinline__ void tdm_load_2d_bf16(
    unsigned ldsOff, const bf16_t* gptr,
    unsigned rowElems, unsigned rowStrideElems,
    unsigned tensorRows, unsigned tileW, unsigned tileH)
{
  unsigned long long ga = (unsigned long long)(uintptr_t)gptr;
  v4u g0;
  g0[0] = 1u;                                               // count=1, user mode
  g0[1] = ldsOff;                                           // LDS byte address
  g0[2] = (unsigned)ga;                                     // global addr lo
  g0[3] = (unsigned)((ga >> 32) & 0x1FFFFFFu) | (2u << 30); // addr hi | type=2
  v8i_t g1;
  g1[0] = (int)(1u << 16);                                  // data_size=1 (2B), wg_mask=0
  g1[1] = (int)((rowElems & 0xFFFFu) << 16);                // tensor_dim0 [15:0]
  g1[2] = (int)(((rowElems >> 16) & 0xFFFFu) |
                ((tensorRows & 0xFFFFu) << 16));            // dim0 hi | tensor_dim1 lo
  g1[3] = (int)(((tensorRows >> 16) & 0xFFFFu) |
                ((tileW & 0xFFFFu) << 16));                 // dim1 hi | tile_dim0
  g1[4] = (int)(tileH & 0xFFFFu);                           // tile_dim1, tile_dim2=0
  g1[5] = (int)rowStrideElems;                              // tensor_dim0_stride lo
  g1[6] = 0;                                                // stride0 hi | stride1 lo
  g1[7] = 0;                                                // stride1 hi
  v4i_t gz = {0, 0, 0, 0};
#if defined(__clang_major__) && (__clang_major__ >= 23)
  v8i_t gz8 = {0, 0, 0, 0, 0, 0, 0, 0};
  __builtin_amdgcn_tensor_load_to_lds(g0, g1, gz, gz, gz8, 0);
#else
  __builtin_amdgcn_tensor_load_to_lds(g0, g1, gz, gz, 0);
#endif
}
#else
#define HAVE_TDM 0
#endif

// ---------------------------------------------------------------------------
// Generic GEMM: C(f32, MxN) = A(f32, MxK) * W(f32, KxN), bf16 WMMA internally.
// Block tile 128x128, k-step 32, 8 waves; wave w owns rows [w*16, w*16+16).
// Double-buffered through registers; 8 B-fragments preloaded per k-step.
// ---------------------------------------------------------------------------
#define GTM 128
#define GTN 128
#define GTK 32
#define LDA_STRIDE 48   // bf16 units (96 B, multiple of 16 B for ds_load_b128)

__global__ __launch_bounds__(256) void gemm_bf16wmma_kernel(
    const float* __restrict__ A, const float* __restrict__ W,
    float* __restrict__ C, int M, int N, int K)
{
  __shared__ bf16_t lA[GTM * LDA_STRIDE];   // [m][k]   12 KB
  __shared__ bf16_t lB[GTN * LDA_STRIDE];   // [n][k]   12 KB (W transposed)

  const int tid      = threadIdx.x;
  const int wave     = tid >> 5;
  const int lane     = tid & 31;
  const int lane16   = lane & 15;
  const int laneHalf = lane >> 4;           // 0 or 1
  const int mBase    = blockIdx.x * GTM;
  const int nBase    = blockIdx.y * GTN;

  v8f acc[8] = {};

  // staging assignments (fixed per thread)
  const int arow = tid >> 1, acol = (tid & 1) * 16;   // A: 16 consecutive f32
  const int wrow = tid >> 3, wcol = (tid & 7) * 16;   // W: 16 consecutive f32
  const float* aPtr = A + (size_t)(mBase + arow) * K + acol;
  const float* wPtr = W + (size_t)wrow * N + nBase + wcol;

  float aReg[16], wReg[16];
  #pragma unroll
  for (int i = 0; i < 16; ++i) aReg[i] = aPtr[i];
  #pragma unroll
  for (int i = 0; i < 16; ++i) wReg[i] = wPtr[i];

  for (int k0 = 0; k0 < K; k0 += GTK) {
    // ---- commit staged registers to LDS (f32 -> bf16) ----
    {
      Frag16 buf;
      #pragma unroll
      for (int i = 0; i < 16; ++i) buf.h[i] = (bf16_t)aReg[i];
      uint4* dst = (uint4*)&lA[arow * LDA_STRIDE + acol];
      dst[0] = buf.q[0];
      dst[1] = buf.q[1];
    }
    #pragma unroll
    for (int i = 0; i < 16; ++i)
      lB[(wcol + i) * LDA_STRIDE + wrow] = (bf16_t)wReg[i];
    __syncthreads();

    // ---- issue next tile's global loads; they overlap the WMMA chain ----
    if (k0 + GTK < K) {
      const float* aN = aPtr + (k0 + GTK);
      const float* wN = wPtr + (size_t)(k0 + GTK) * N;
      #pragma unroll
      for (int i = 0; i < 16; ++i) aReg[i] = aN[i];
      #pragma unroll
      for (int i = 0; i < 16; ++i) wReg[i] = wN[i];
    }

    // ---- A fragment (ISA layout: lane<16 K0-7/16-23, lane>=16 K8-15/24-31) ----
    Frag16 af;
    {
      int off = (wave * 16 + lane16) * LDA_STRIDE + laneHalf * 8;
      af.q[0] = *(const uint4*)&lA[off];
      af.q[1] = *(const uint4*)&lA[off + 16];
    }
    // ---- preload all 8 B fragments, then chain WMMAs ----
    Frag16 bfr[8];
    #pragma unroll
    for (int t = 0; t < 8; ++t) {
      int off = (t * 16 + lane16) * LDA_STRIDE + laneHalf * 8;
      bfr[t].q[0] = *(const uint4*)&lB[off];
      bfr[t].q[1] = *(const uint4*)&lB[off + 16];
    }
    #pragma unroll
    for (int t = 0; t < 8; ++t)
      acc[t] = __builtin_amdgcn_wmma_f32_16x16x32_bf16(
          false, af.v, false, bfr[t].v, (short)0, acc[t], false, false);
    __syncthreads();
  }

  // ---- epilogue: C/D layout — VGPR r = row r (lanes 0-15) / row 8+r (16-31)
  int rowBase = mBase + wave * 16 + laneHalf * 8;
  #pragma unroll
  for (int t = 0; t < 8; ++t) {
    int col = nBase + t * 16 + lane16;
    #pragma unroll
    for (int r = 0; r < 8; ++r)
      C[(size_t)(rowBase + r) * N + col] = acc[t][r];
  }
}

// ---------------------------------------------------------------------------
// RoPE + pack: src f32 [B*S, nH*D] -> dst bf16 [B, nH, S, D]
// ---------------------------------------------------------------------------
__global__ __launch_bounds__(256) void rope_pack_kernel(
    const float* __restrict__ src, bf16_t* __restrict__ dst,
    int nH, int applyRope)
{
  int idx = blockIdx.x * blockDim.x + threadIdx.x;
  int d = idx % HEADDIM;
  int s = (idx / HEADDIM) % SEQ;
  int h = (idx / (HEADDIM * SEQ)) % nH;
  int b = idx / (HEADDIM * SEQ * nH);

  const float* row = src + (size_t)(b * SEQ + s) * (nH * HEADDIM) + h * HEADDIM;
  float x = row[d];
  float out = x;
  if (applyRope) {
    const int half = HEADDIM / 2;
    int i = (d < half) ? d : (d - half);
    // inv_freq = theta^(-2i/D); log2(10000) = 13.2877124
    float angle = (float)s * exp2f(-(2.0f * (float)i / (float)HEADDIM) * 13.287712379549449f);
    float partner = (d < half) ? -row[2 * i + 1] : row[2 * i];
    out = x * cosf(angle) + partner * sinf(angle);
  }
  dst[(size_t)((b * nH + h) * SEQ + s) * HEADDIM + d] = (bf16_t)out;
}

// ---------------------------------------------------------------------------
// Flash attention: grid (B*H, S/128); 8 waves, wave w owns query rows
// [y*128 + w*16, +16). K tiles double-buffered in LDS via issue-ahead TDM
// (async tensor DMA overlaps WMMA/softmax); V tiles double-buffered through
// registers and committed transposed (TDM cannot transpose).
// ---------------------------------------------------------------------------
#define KT 32
#define PSTRIDE 48
#define KBUFSZ (KT * HEADDIM)        // bf16 elems per K buffer
#define VBUFSZ (HEADDIM * PSTRIDE)   // bf16 elems per V^T buffer

__global__ __launch_bounds__(256) void flash_attn_kernel(
    const bf16_t* __restrict__ Q, const bf16_t* __restrict__ Kc,
    const bf16_t* __restrict__ V, const float* __restrict__ amask,
    float* __restrict__ Out)
{
  __shared__ bf16_t lK[2 * KBUFSZ];          // [buf][key][d]   16 KB
  __shared__ bf16_t lV[2 * VBUFSZ];          // [buf][d][key]   24 KB
  __shared__ bf16_t lP[8 * 16 * PSTRIDE];    // per-wave P      12 KB

  const int tid      = threadIdx.x;
  const int wave     = tid >> 5;
  const int lane     = tid & 31;
  const int lane16   = lane & 15;
  const int laneHalf = lane >> 4;

  const int h   = blockIdx.x % NHEADS;
  const int b   = blockIdx.x / NHEADS;
  const int kvh = h / (NHEADS / NKVHEADS);
  const int qBase = blockIdx.y * 128 + wave * 16;

  const bf16_t* Qhead = Q  + (size_t)(b * NHEADS   + h)   * SEQ * HEADDIM;
  const bf16_t* Khead = Kc + (size_t)(b * NKVHEADS + kvh) * SEQ * HEADDIM;
  const bf16_t* Vhead = V  + (size_t)(b * NKVHEADS + kvh) * SEQ * HEADDIM;
  const float*  maskRow = amask + (size_t)b * SEQ;

  // Q fragments, 4 chunks across head dim (loaded once)
  Frag16 qf[4];
  {
    const bf16_t* qr = Qhead + (size_t)(qBase + lane16) * HEADDIM;
    #pragma unroll
    for (int c = 0; c < 4; ++c) {
      int off = c * 32 + laneHalf * 8;
      qf[c].q[0] = *(const uint4*)(qr + off);
      qf[c].q[1] = *(const uint4*)(qr + off + 16);
    }
  }

  v8f oacc[8] = {};
  float mrun[8], lrun[8];
  #pragma unroll
  for (int r = 0; r < 8; ++r) { mrun[r] = -3.0e38f; lrun[r] = 0.0f; }

  const float scale = 0.08838834764831845f;    // 1/sqrt(128)
  const int qMax = qBase + 15;
  const int nkb  = (blockIdx.y + 1) * (128 / KT);   // causal key-block limit

  // ---- pipeline prologue: first V tile into regs; first K tile via TDM ----
  const int vkey = tid >> 3;
  const int vdC  = (tid & 7) * 16;
  uint4 vReg0, vReg1;
  {
    const bf16_t* src = Vhead + (size_t)vkey * HEADDIM + vdC;
    vReg0 = *(const uint4*)src;
    vReg1 = *(const uint4*)(src + 8);
  }
#if HAVE_TDM
  if (wave == 0) {
    tdm_load_2d_bf16((unsigned)(uintptr_t)&lK[0], Khead,
                     HEADDIM, HEADDIM, (unsigned)SEQ, HEADDIM, KT);
  }
#endif

  for (int kb = 0; kb < nkb; ++kb) {
    const int kbase = kb * KT;
    const int cur   = kb & 1;
    bf16_t* lKc = &lK[cur * KBUFSZ];
    bf16_t* lVc = &lV[cur * VBUFSZ];

    // ---- commit staged V registers, transposed, to current buffer ----
    {
      Frag16 vb;
      vb.q[0] = vReg0;
      vb.q[1] = vReg1;
      #pragma unroll
      for (int i = 0; i < 16; ++i)
        lVc[(vdC + i) * PSTRIDE + vkey] = vb.h[i];
    }
    // ---- K tile: issue-ahead TDM into alternate buffer, then make sure the
    //      current tile's DMA has landed (in-order TDM completion per wave) ----
#if HAVE_TDM
    if (wave == 0) {
      if (kb + 1 < nkb) {
        tdm_load_2d_bf16((unsigned)(uintptr_t)&lK[(1 - cur) * KBUFSZ],
                         Khead + (size_t)(kbase + KT) * HEADDIM,
                         HEADDIM, HEADDIM, (unsigned)(SEQ - kbase - KT),
                         HEADDIM, KT);
        __builtin_amdgcn_s_wait_tensorcnt(1);   // current tile done, next in flight
      } else {
        __builtin_amdgcn_s_wait_tensorcnt(0);
      }
    }
#else
    {
      const bf16_t* src = Khead + (size_t)(kbase + vkey) * HEADDIM + vdC;
      uint4 a0 = *(const uint4*)src;
      uint4 a1 = *(const uint4*)(src + 8);
      uint4* dst = (uint4*)&lKc[vkey * HEADDIM + vdC];
      dst[0] = a0; dst[1] = a1;
    }
#endif
    __syncthreads();

    // ---- prefetch next V tile into regs (overlaps compute below) ----
    if (kb + 1 < nkb) {
      const bf16_t* src = Vhead + (size_t)(kbase + KT + vkey) * HEADDIM + vdC;
      vReg0 = *(const uint4*)src;
      vReg1 = *(const uint4*)(src + 8);
    }

    if (kbase <= qMax) {
      // ---- scores: preload 8 K fragments, then two 16x16 tiles over 4 chunks
      Frag16 kf[8];
      #pragma unroll
      for (int c = 0; c < 4; ++c) {
        int o0 = lane16 * HEADDIM + c * 32 + laneHalf * 8;
        kf[2 * c].q[0] = *(const uint4*)&lKc[o0];
        kf[2 * c].q[1] = *(const uint4*)&lKc[o0 + 16];
        int o1 = (16 + lane16) * HEADDIM + c * 32 + laneHalf * 8;
        kf[2 * c + 1].q[0] = *(const uint4*)&lKc[o1];
        kf[2 * c + 1].q[1] = *(const uint4*)&lKc[o1 + 16];
      }
      v8f sc0 = {}, sc1 = {};
      #pragma unroll
      for (int c = 0; c < 4; ++c) {
        sc0 = __builtin_amdgcn_wmma_f32_16x16x32_bf16(false, qf[c].v, false, kf[2 * c].v,     (short)0, sc0, false, false);
        sc1 = __builtin_amdgcn_wmma_f32_16x16x32_bf16(false, qf[c].v, false, kf[2 * c + 1].v, (short)0, sc1, false, false);
      }

      // ---- causal mask + online softmax (rows live in VGPR index r) ----
      int qRow = qBase + laneHalf * 8;
      int key0 = kbase + lane16;
      int key1 = key0 + 16;
      float am0 = maskRow[key0];
      float am1 = maskRow[key1];
      float p0[8], p1[8], alpha[8];
      #pragma unroll
      for (int r = 0; r < 8; ++r) {
        int q = qRow + r;
        float s0 = (key0 <= q) ? (sc0[r] * scale + am0) : -3.0e38f;
        float s1 = (key1 <= q) ? (sc1[r] * scale + am1) : -3.0e38f;
        float mv = fmaxf(s0, s1);
        #pragma unroll
        for (int o = 1; o < 16; o <<= 1)
          mv = fmaxf(mv, __shfl_xor(mv, o, 16));   // row max within lane half
        float mnew = fmaxf(mrun[r], mv);
        float a  = __expf(mrun[r] - mnew);
        float e0 = __expf(s0 - mnew);
        float e1 = __expf(s1 - mnew);
        float rs = e0 + e1;
        #pragma unroll
        for (int o = 1; o < 16; o <<= 1)
          rs += __shfl_xor(rs, o, 16);             // row sum
        lrun[r] = lrun[r] * a + rs;
        mrun[r] = mnew;
        alpha[r] = a;
        p0[r] = e0; p1[r] = e1;
      }
      #pragma unroll
      for (int t = 0; t < 8; ++t) {
        #pragma unroll
        for (int r = 0; r < 8; ++r) oacc[t][r] *= alpha[r];
      }

      // ---- repack P (C-layout -> A-layout) through per-wave LDS ----
      bf16_t* pw = &lP[wave * 16 * PSTRIDE];
      #pragma unroll
      for (int r = 0; r < 8; ++r) {
        int prow = laneHalf * 8 + r;
        pw[prow * PSTRIDE + lane16]      = (bf16_t)p0[r];
        pw[prow * PSTRIDE + 16 + lane16] = (bf16_t)p1[r];
      }
      Frag16 pf;
      {
        int off = lane16 * PSTRIDE + laneHalf * 8;
        pf.q[0] = *(const uint4*)&pw[off];
        pf.q[1] = *(const uint4*)&pw[off + 16];
      }
      // ---- PV: preload 8 V fragments, then 8 WMMAs over d-chunks ----
      Frag16 vf[8];
      #pragma unroll
      for (int t = 0; t < 8; ++t) {
        int off = (t * 16 + lane16) * PSTRIDE + laneHalf * 8;
        vf[t].q[0] = *(const uint4*)&lVc[off];
        vf[t].q[1] = *(const uint4*)&lVc[off + 16];
      }
      #pragma unroll
      for (int t = 0; t < 8; ++t)
        oacc[t] = __builtin_amdgcn_wmma_f32_16x16x32_bf16(false, pf.v, false, vf[t].v, (short)0, oacc[t], false, false);
    }
    __syncthreads();
  }

  // ---- finalize: Out [B, S, H, D] (row-major (B*S) x (H*D) for O-proj) ----
  #pragma unroll
  for (int r = 0; r < 8; ++r) {
    int q = qBase + laneHalf * 8 + r;
    float inv = 1.0f / lrun[r];
    float* orow = Out + ((size_t)(b * SEQ + q) * NHEADS + h) * HEADDIM;
    #pragma unroll
    for (int t = 0; t < 8; ++t)
      orow[t * 16 + lane16] = oacc[t][r] * inv;
  }
}

// ---------------------------------------------------------------------------
extern "C" void kernel_launch(void* const* d_in, const int* in_sizes, int n_in,
                              void* d_out, int out_size, void* d_ws, size_t ws_size,
                              hipStream_t stream)
{
  (void)in_sizes; (void)n_in; (void)out_size; (void)ws_size;
  const float* hidden = (const float*)d_in[0];
  const float* amask  = (const float*)d_in[1];
  const float* Wq     = (const float*)d_in[2];
  const float* Wk     = (const float*)d_in[3];
  const float* Wv     = (const float*)d_in[4];
  const float* Wo     = (const float*)d_in[5];
  float* out = (float*)d_out;

  char* ws = (char*)d_ws;
  size_t off = 0;
  auto alloc = [&](size_t bytes) -> char* {
    char* p = ws + off;
    off += (bytes + 255) & ~(size_t)255;
    return p;
  };

  float*  qf32 = (float*)alloc((size_t)MROWS * 2048 * sizeof(float)); // reused as attn out
  float*  kf32 = (float*)alloc((size_t)MROWS * 512  * sizeof(float));
  float*  vf32 = (float*)alloc((size_t)MROWS * 512  * sizeof(float));
  bf16_t* Qb   = (bf16_t*)alloc((size_t)MROWS * 2048 * sizeof(bf16_t));
  bf16_t* Kb   = (bf16_t*)alloc((size_t)MROWS * 512  * sizeof(bf16_t));
  bf16_t* Vb   = (bf16_t*)alloc((size_t)MROWS * 512  * sizeof(bf16_t));

  dim3 blk(256);

  // QKV projections
  gemm_bf16wmma_kernel<<<dim3(MROWS / 128, 2048 / 128), blk, 0, stream>>>(hidden, Wq, qf32, MROWS, 2048, HIDDEN);
  gemm_bf16wmma_kernel<<<dim3(MROWS / 128,  512 / 128), blk, 0, stream>>>(hidden, Wk, kf32, MROWS,  512, HIDDEN);
  gemm_bf16wmma_kernel<<<dim3(MROWS / 128,  512 / 128), blk, 0, stream>>>(hidden, Wv, vf32, MROWS,  512, HIDDEN);

  // RoPE + pack to [B, H, S, D] bf16
  rope_pack_kernel<<<(BATCH * SEQ * NHEADS   * HEADDIM) / 256, blk, 0, stream>>>(qf32, Qb, NHEADS, 1);
  rope_pack_kernel<<<(BATCH * SEQ * NKVHEADS * HEADDIM) / 256, blk, 0, stream>>>(kf32, Kb, NKVHEADS, 1);
  rope_pack_kernel<<<(BATCH * SEQ * NKVHEADS * HEADDIM) / 256, blk, 0, stream>>>(vf32, Vb, NKVHEADS, 0);

  // Flash attention -> attn output [B, S, H, D] f32 (reuses qf32 scratch)
  float* attn = qf32;
  flash_attn_kernel<<<dim3(BATCH * NHEADS, SEQ / 128), blk, 0, stream>>>(Qb, Kb, Vb, amask, attn);

  // Output projection
  gemm_bf16wmma_kernel<<<dim3(MROWS / 128, 2048 / 128), blk, 0, stream>>>(attn, Wo, out, MROWS, 2048, HIDDEN);
}
